// MemoryANN_56942676411042
// MI455X (gfx1250) — matrix operand order repaired and checked by reading, not compiled, not run on hardware
//
#include <hip/hip_runtime.h>
#include <hip/hip_bf16.h>

typedef __attribute__((ext_vector_type(16))) _Float16     v16h;
typedef __attribute__((ext_vector_type(8)))  float        v8f;
typedef __attribute__((ext_vector_type(8)))  unsigned int v8u;

#define N_ACTIONS 4
#define TFULL   1024
#define TSTEPS  1023          // T-1 output steps
#define FORGETF 0.05f

// Branch-free tanh: CDNA5 hardware V_TANH_F32 (TRANS32 op, co-executes with
// VALU) when available; otherwise a branch-free exp-based fallback.
__device__ __forceinline__ float tanh_fast(float x) {
#if __has_builtin(__builtin_amdgcn_tanhf)
  return __builtin_amdgcn_tanhf(x);
#else
  float e2 = __expf(2.0f * x);                       // v_exp_f32 path
  return (e2 - 1.0f) * __builtin_amdgcn_rcpf(e2 + 1.0f);
#endif
}

// Convert state held in two f32 D-layout accumulators
// (dlo: rows j=0..15, dhi: j=16..31; lane holds (j = v+8*hi, m = lane%16))
// into f16 B operands for v_wmma_f32_16x16x32_f16, in the contiguous K order the A weights use:
//   lane l: N = l%16, element i holds K = 16*(l>=16) + i.
// tanh -> pack f16 pairs -> 4 packed cross-half shuffles per matrix.
__device__ __forceinline__ v16h state_to_b(v8f dlo, v8f dhi, int hi) {
  float tl[8], th[8];
#pragma unroll
  for (int v = 0; v < 8; ++v) {
    tl[v] = tanh_fast(dlo[v]);   // state row j = v + 8*hi      (this lane)
    th[v] = tanh_fast(dhi[v]);   // state row j = 16 + v + 8*hi (this lane)
  }
  unsigned int pk0[4], pk1[4];   // packed f16 pairs (rows 2p,2p+1 of each acc)
#pragma unroll
  for (int p = 0; p < 4; ++p) {
    pk0[p] = __builtin_bit_cast(unsigned int,
               __builtin_amdgcn_cvt_pkrtz(tl[2 * p], tl[2 * p + 1]));
    pk1[p] = __builtin_bit_cast(unsigned int,
               __builtin_amdgcn_cvt_pkrtz(th[2 * p], th[2 * p + 1]));
  }
  v8u words;
#pragma unroll
  for (int p = 0; p < 4; ++p) {
    // low lanes need tl-words from the high partner; high lanes need th-words
    // from the low partner.
    unsigned int send = hi ? pk0[p] : pk1[p];
    unsigned int recv = (unsigned int)__shfl_xor((int)send, 16, 32);
    words[p]     = hi ? recv   : pk0[p];  // K = 16*hi + {2p,2p+1}   (from low lane)
    words[4 + p] = hi ? pk1[p] : recv;    // K = 16*hi + 8 + {2p,..} (from high lane)
  }
  return __builtin_bit_cast(v16h, words);
}

__global__ __launch_bounds__(32) void mann_rnn_wmma(
    const int*   __restrict__ actions,  // [B, T]
    const float* __restrict__ rewards,  // [B, T]
    const float* __restrict__ w_r1,     // [32, 33]
    const float* __restrict__ b_r1,     // [32]
    const float* __restrict__ w_r2,     // [1, 32]
    const float* __restrict__ b_r2,     // [1]
    const float* __restrict__ w_a1,     // [32, 36]
    const float* __restrict__ b_a1,     // [32]
    const float* __restrict__ w_a2,     // [4, 32]
    const float* __restrict__ b_a2,     // [4]
    float*       __restrict__ out)      // [B, T-1, 4]
{
  const int lane = threadIdx.x;   // wave32
  const int lm   = lane & 15;
  const int hi   = lane >> 4;     // 0 = low half, 1 = high half
  const int b0   = blockIdx.x * 16;
  const int brow = b0 + lm;       // batch row for this lane column

  // ---- resident weights as WMMA A operands (f16) ----
  // A element i -> K = 16*hi + i ; A row M = lane%16.
  v16h a_r1_0, a_r1_1, a_a1_0, a_a1_1, a_r2, a_a2;
#pragma unroll
  for (int i = 0; i < 16; ++i) {
    const int K = 16 * hi + i;
    a_r1_0[i] = (_Float16)w_r1[(0  + lm) * 33 + 1 + K];   // hidden part, rows 0..15
    a_r1_1[i] = (_Float16)w_r1[(16 + lm) * 33 + 1 + K];   // rows 16..31
    a_a1_0[i] = (_Float16)w_a1[(0  + lm) * 36 + 4 + K];
    a_a1_1[i] = (_Float16)w_a1[(16 + lm) * 36 + 4 + K];
    a_r2[i]   = (lm == 0) ? (_Float16)w_r2[K]            : (_Float16)0.0f;
    a_a2[i]   = (lm <  4) ? (_Float16)w_a2[lm * 32 + K]  : (_Float16)0.0f;
  }

  // ---- per-lane bias / folded-input constants (C-accumulator layout:
  //      element v -> out row j = 16*hb + v + 8*hi, column m = lane%16) ----
  float br1v[2][8], wr1c0[2][8], ba1v[2][8], wa1c[2][4][8];
#pragma unroll
  for (int hb = 0; hb < 2; ++hb)
#pragma unroll
    for (int v = 0; v < 8; ++v) {
      const int j = 16 * hb + v + 8 * hi;
      br1v[hb][v]  = b_r1[j];
      wr1c0[hb][v] = w_r1[j * 33 + 0];    // reward-input column
      ba1v[hb][v]  = b_a1[j];
#pragma unroll
      for (int s = 0; s < N_ACTIONS; ++s) wa1c[hb][s][v] = w_a1[j * 36 + s];
    }
  v8f c_q = {}, c_c = {};
  if (hi == 0) {
    c_q[0] = b_r2[0];                     // row 0 of padded w_r2 head
#pragma unroll
    for (int v = 0; v < 4; ++v) c_c[v] = b_a2[v];
  }

  // ---- recurrent state ----
  v16h b_sr = {};                         // sr state as f16 B operand (zeros)
  v16h b_sa = {};                         // sa state as f16 B operand
  float q[4] = {0.f, 0.f, 0.f, 0.f};     // Q values (valid on low half)

  const size_t inrow = (size_t)brow * TFULL;

  for (int t = 0; t < TSTEPS; ++t) {
    const float r = rewards[inrow + t];
    const int   a = actions[inrow + t];
    __builtin_prefetch(&rewards[inrow + t + 64], 0, 1);
    __builtin_prefetch(&actions[inrow + t + 64], 0, 1);

    // per-lane one-hot as floats (branch-free: v_cmp + v_cndmask each)
    float oh[N_ACTIONS];
#pragma unroll
    for (int s = 0; s < N_ACTIONS; ++s) oh[s] = (a == s) ? 1.0f : 0.0f;

    // ---- reward module layer 1: W_r1h * sr + (b_r1 + r*w_r1[:,0]) ----
    v8f cr0, cr1;
#pragma unroll
    for (int v = 0; v < 8; ++v) {
      cr0[v] = fmaf(r, wr1c0[0][v], br1v[0][v]);
      cr1[v] = fmaf(r, wr1c0[1][v], br1v[1][v]);
    }
    v8f dr0 = __builtin_amdgcn_wmma_f32_16x16x32_f16(false, a_r1_0, false, b_sr, (short)0, cr0, false, false);
    v8f dr1 = __builtin_amdgcn_wmma_f32_16x16x32_f16(false, a_r1_1, false, b_sr, (short)0, cr1, false, false);
    b_sr = state_to_b(dr0, dr1, hi);      // tanh + relayout -> new sr state

    // ---- reward module layer 2: q_new = w_r2 . sr_t + b_r2 (row 0 of D) ----
    v8f dq = __builtin_amdgcn_wmma_f32_16x16x32_f16(false, a_r2, false, b_sr, (short)0, c_q, false, false);
    const float q_new = dq[0];            // valid on low-half lanes (row M=0)

    // ---- action module layer 1: one-hot column folded into C via FMA chain
    //      (branch-free; compiler packs into v_pk_fma_f32) ----
    v8f ca0, ca1;
#pragma unroll
    for (int v = 0; v < 8; ++v) {
      float w0 = ba1v[0][v];
      float w1 = ba1v[1][v];
#pragma unroll
      for (int s = 0; s < N_ACTIONS; ++s) {
        w0 = fmaf(oh[s], wa1c[0][s][v], w0);
        w1 = fmaf(oh[s], wa1c[1][s][v], w1);
      }
      ca0[v] = w0;
      ca1[v] = w1;
    }
    v8f da0 = __builtin_amdgcn_wmma_f32_16x16x32_f16(false, a_a1_0, false, b_sa, (short)0, ca0, false, false);
    v8f da1 = __builtin_amdgcn_wmma_f32_16x16x32_f16(false, a_a1_1, false, b_sa, (short)0, ca1, false, false);
    b_sa = state_to_b(da0, da1, hi);      // tanh + relayout -> new sa state

    // ---- action module layer 2: c_t rows 0..3 ----
    v8f dc = __builtin_amdgcn_wmma_f32_16x16x32_f16(false, a_a2, false, b_sa, (short)0, c_c, false, false);

    // ---- Q decay + chosen-action replace (branch-free blend), logits, store ----
    float4 lg;
#pragma unroll
    for (int n = 0; n < N_ACTIONS; ++n) {
      const float qd = q[n] * (1.0f - FORGETF);       // Q_INIT == 0
      q[n] = fmaf(oh[n], q_new - qd, qd);             // oh ? q_new : qd
      (&lg.x)[n] = q[n] + dc[n];
    }
    if (hi == 0) {
      *reinterpret_cast<float4*>(out + ((size_t)brow * TSTEPS + t) * 4) = lg;
    }
  }
}

extern "C" void kernel_launch(void* const* d_in, const int* in_sizes, int n_in,
                              void* d_out, int out_size, void* d_ws, size_t ws_size,
                              hipStream_t stream) {
  (void)in_sizes; (void)n_in; (void)d_ws; (void)ws_size; (void)out_size;
  const int*   actions = (const int*)  d_in[0];
  const float* rewards = (const float*)d_in[1];
  const float* w_r1    = (const float*)d_in[2];
  const float* b_r1    = (const float*)d_in[3];
  const float* w_r2    = (const float*)d_in[4];
  const float* b_r2    = (const float*)d_in[5];
  const float* w_a1    = (const float*)d_in[6];
  const float* b_a1    = (const float*)d_in[7];
  const float* w_a2    = (const float*)d_in[8];
  const float* b_a2    = (const float*)d_in[9];
  float* out = (float*)d_out;

  // B = 8192 rows, 16 rows per wave -> 512 single-wave workgroups,
  // each an independent 1023-step recurrence (pipelines across SIMDs).
  dim3 grid(8192 / 16), block(32);
  mann_rnn_wmma<<<grid, block, 0, stream>>>(actions, rewards,
                                            w_r1, b_r1, w_r2, b_r2,
                                            w_a1, b_a1, w_a2, b_a2, out);
}
